// InteractModel_29669634080803
// MI455X (gfx1250) — matrix operand order repaired and checked by reading, not compiled, not run on hardware
//
#include <hip/hip_runtime.h>
#include <hip/hip_bf16.h>

// ---------------------------------------------------------------------------
// InteractModel for MI455X (gfx1250).
// Heavy GEMMs: v_wmma_f32_16x16x32_f16, 64x64 block tile, 4 waves (2x2),
// each wave owns a 32x32 output (4 accumulators -> 4 WMMAs per K-step).
// Tile staging: vectorized global_load_b128 + ds_store_b64 fast paths
// (f32 -> f16 conversion in flight) + global_prefetch of the next K tile;
// guarded scalar path only on edge tiles.
// Compute-bound (~210 GFLOP vs ~1.5 GB @ 23.3 TB/s): matrix pipe is the target.
// Weight matrices (<3 MB) stay resident in the 192 MB L2 across M-tiles.
// ---------------------------------------------------------------------------

#define DM      768
#define NB      16          // text batch
#define SEQ     201
#define NHD     4           // heads
#define HD      192         // head dim
#define NNODES  50000
#define NEDGES  150000
#define KNBR    20
#define ADIM    16
#define SCP     224         // padded row stride for text attention scores
#define SUBSCP  32          // padded row stride for neighborhood scores

typedef __attribute__((ext_vector_type(16))) _Float16 v16h;
typedef __attribute__((ext_vector_type(8)))  _Float16 v8h;
typedef __attribute__((ext_vector_type(4)))  _Float16 v4h;
typedef __attribute__((ext_vector_type(8)))  float    v8f;

// ---------------------------------------------------------------------------
// Batched WMMA GEMM:  C = act(A @ B(^T) + bias)
//   block = 128 threads (4 waves); block tile 64(M) x 64(N), K-step 32.
//   wave (2x2 grid) computes 32x32 via 2x2 v_wmma_f32_16x16x32_f16.
//   batch z -> (b = z/nhdiv, h = z%nhdiv) with independent strides.
// ---------------------------------------------------------------------------
__global__ __launch_bounds__(128) void wmma_gemm_kernel(
    const float* __restrict__ A, long lda, long sA1, long sA2,
    const float* __restrict__ B, long ldb, long sB1, long sB2, int transB,
    float* __restrict__ C, long ldc, long sC1, long sC2,
    int M, int N, int K,
    const float* __restrict__ bias, int act, int nhdiv)
{
    __shared__ __align__(16) _Float16 As[64][40];   // 64 x 32 (+pad)
    __shared__ __align__(16) _Float16 Bs[32][72];   // 32 x 64 (+pad)

    const int z  = blockIdx.z;
    const int bz = z / nhdiv, hz = z % nhdiv;
    A += (long)bz * sA1 + (long)hz * sA2;
    B += (long)bz * sB1 + (long)hz * sB2;
    C += (long)bz * sC1 + (long)hz * sC2;

    const int m0  = blockIdx.y * 64;
    const int n0  = blockIdx.x * 64;
    const int tid = threadIdx.x;
    const int lane = tid & 31;
    const int wave = tid >> 5;
    const int wm = (wave >> 1) * 32;   // wave M offset in tile
    const int wn = (wave & 1) * 32;    // wave N offset in tile

    const bool alignedA = ((((unsigned long long)A) & 15) == 0) && ((lda & 3) == 0);
    const bool alignedB = ((((unsigned long long)B) & 15) == 0) && ((ldb & 3) == 0);
    const bool fullM = (m0 + 64 <= M);
    const bool fullN = (n0 + 64 <= N);

    v8f acc[2][2] = {};

    for (int k0 = 0; k0 < K; k0 += 32) {
        const bool fullK = (k0 + 32 <= K);
        const bool moreK = (k0 + 64 <= K);   // next tile fully present

        // ---- stage A tile (64 x 32) ----
        if (alignedA && fullK && fullM) {
            const int rb = tid >> 3;            // 0..15
            const int c  = (tid & 7) * 4;       // 0..28
            const float* ap = A + (long)(m0 + rb) * lda + k0 + c;
#pragma unroll
            for (int t = 0; t < 4; ++t) {
                float4 v = *(const float4*)(ap + (long)16 * t * lda);
                v4h h; h[0] = (_Float16)v.x; h[1] = (_Float16)v.y;
                       h[2] = (_Float16)v.z; h[3] = (_Float16)v.w;
                *(v4h*)&As[rb + 16 * t][c] = h;
            }
            if (moreK) __builtin_prefetch(ap + 32, 0, 1);   // next K tile
        } else {
            for (int i = tid; i < 64 * 32; i += 128) {
                int r = i >> 5, c = i & 31;
                int gm = m0 + r, gk = k0 + c;
                float v = (gm < M && gk < K) ? A[(long)gm * lda + gk] : 0.f;
                As[r][c] = (_Float16)v;
            }
        }

        // ---- stage B tile (32 x 64) ----
        if (!transB) {
            if (alignedB && fullK && fullN) {
                const int rb = tid >> 4;          // 0..7
                const int c  = (tid & 15) * 4;    // 0..60
                const float* bp = B + (long)(k0 + rb) * ldb + n0 + c;
#pragma unroll
                for (int t = 0; t < 4; ++t) {
                    float4 v = *(const float4*)(bp + (long)8 * t * ldb);
                    v4h h; h[0] = (_Float16)v.x; h[1] = (_Float16)v.y;
                           h[2] = (_Float16)v.z; h[3] = (_Float16)v.w;
                    *(v4h*)&Bs[rb + 8 * t][c] = h;
                }
                if (moreK) __builtin_prefetch(bp + 32 * ldb, 0, 1);
            } else {
                for (int i = tid; i < 32 * 64; i += 128) {
                    int r = i >> 6, c = i & 63;
                    int gk = k0 + r, gn = n0 + c;
                    float v = (gk < K && gn < N) ? B[(long)gk * ldb + gn] : 0.f;
                    Bs[r][c] = (_Float16)v;
                }
            }
        } else {
            if (alignedB && fullK && fullN) {
                const int kb4 = (tid & 7) * 4;    // 0..28 along K
                const int cb  = tid >> 3;         // 0..15 along N
#pragma unroll
                for (int t = 0; t < 4; ++t) {
                    int c = cb + 16 * t;
                    float4 v = *(const float4*)(B + (long)(n0 + c) * ldb + k0 + kb4);
                    Bs[kb4 + 0][c] = (_Float16)v.x;
                    Bs[kb4 + 1][c] = (_Float16)v.y;
                    Bs[kb4 + 2][c] = (_Float16)v.z;
                    Bs[kb4 + 3][c] = (_Float16)v.w;
                }
                if (moreK) __builtin_prefetch(B + (long)(n0 + cb) * ldb + k0 + 32 + kb4, 0, 1);
            } else {
                for (int i = tid; i < 32 * 64; i += 128) {
                    int r = i >> 6, c = i & 63;
                    int gk = k0 + r, gn = n0 + c;
                    float v = (gk < K && gn < N) ? B[(long)gn * ldb + gk] : 0.f;
                    Bs[r][c] = (_Float16)v;
                }
            }
        }
        __syncthreads();

        // ---- fragments ----
        // A: lane = M row (l&15), k_base = (l<16?0:8);
        //    halves 0..7 -> K=kb..kb+7, halves 8..15 -> K=kb+16..kb+23
        const int kb = (lane < 16) ? 0 : 8;
        v16h af[2], bf[2];
#pragma unroll
        for (int mi = 0; mi < 2; ++mi) {
            const int arow = wm + mi * 16 + (lane & 15);
            v8h lo = *(const v8h*)&As[arow][kb];
            v8h hi = *(const v8h*)&As[arow][kb + 16];
#pragma unroll
            for (int j = 0; j < 8; ++j) { af[mi][j] = lo[j]; af[mi][8 + j] = hi[j]; }
        }
        // B: lane = N column (l&15); halves 0..7 -> K=kb..kb+7, halves 8..15 -> K=kb+16..kb+23
#pragma unroll
        for (int ni = 0; ni < 2; ++ni) {
            const int cb = wn + ni * 16 + (lane & 15);
#pragma unroll
            for (int j = 0; j < 8; ++j) { bf[ni][j] = Bs[kb + j][cb]; bf[ni][8 + j] = Bs[kb + 16 + j][cb]; }
        }

#pragma unroll
        for (int mi = 0; mi < 2; ++mi)
#pragma unroll
            for (int ni = 0; ni < 2; ++ni)
                acc[mi][ni] = __builtin_amdgcn_wmma_f32_16x16x32_f16(
                    false, af[mi], false, bf[ni], (short)0, acc[mi][ni], false, false);
        __syncthreads();
    }

    // ---- store: C/D layout lane -> N=(l&15); VGPR r -> M=r | 8+r ----
#pragma unroll
    for (int mi = 0; mi < 2; ++mi) {
#pragma unroll
        for (int ni = 0; ni < 2; ++ni) {
            const int cn = n0 + wn + ni * 16 + (lane & 15);
            float bv = 0.f;
            if (bias && cn < N) bv = bias[cn];
#pragma unroll
            for (int r = 0; r < 8; ++r) {
                int cm = m0 + wm + mi * 16 + ((lane < 16) ? r : (8 + r));
                if (cm < M && cn < N) {
                    float v = acc[mi][ni][r] + bv;
                    if (act == 1) v = (v > 0.f) ? v : 0.01f * v;   // leaky_relu
                    C[(long)cm * ldc + cn] = v;
                }
            }
        }
    }
}

// ---------------------------------------------------------------------------
// Small-K / small-N scalar GEMM (K=3,5 projections; final 768->2 head)
// ---------------------------------------------------------------------------
__global__ void smallk_gemm_kernel(const float* __restrict__ A, int lda,
                                   const float* __restrict__ W,
                                   const float* __restrict__ bias,
                                   float* __restrict__ C, long ldc,
                                   int M, int N, int K)
{
    long i = blockIdx.x * 256L + threadIdx.x;
    if (i >= (long)M * N) return;
    int m = (int)(i / N), n = (int)(i % N);
    float s = bias ? bias[n] : 0.f;
    for (int k = 0; k < K; ++k) s += A[(long)m * lda + k] * W[(long)k * N + n];
    C[(long)m * ldc + n] = s;
}

__global__ __launch_bounds__(256) void layernorm_kernel(
    const float* __restrict__ x, const float* __restrict__ g,
    const float* __restrict__ b, float* __restrict__ y, int nc)
{
    __shared__ float red[256];
    long row = blockIdx.x;
    const float* xr = x + row * nc;
    float s = 0.f;
    for (int c = threadIdx.x; c < nc; c += 256) s += xr[c];
    red[threadIdx.x] = s; __syncthreads();
    for (int o = 128; o > 0; o >>= 1) {
        if (threadIdx.x < o) red[threadIdx.x] += red[threadIdx.x + o];
        __syncthreads();
    }
    float mean = red[0] / nc; __syncthreads();
    float v = 0.f;
    for (int c = threadIdx.x; c < nc; c += 256) { float d = xr[c] - mean; v += d * d; }
    red[threadIdx.x] = v; __syncthreads();
    for (int o = 128; o > 0; o >>= 1) {
        if (threadIdx.x < o) red[threadIdx.x] += red[threadIdx.x + o];
        __syncthreads();
    }
    float rstd = rsqrtf(red[0] / nc + 1e-5f);
    float* yr = y + row * nc;
    for (int c = threadIdx.x; c < nc; c += 256)
        yr[c] = (xr[c] - mean) * rstd * g[c] + b[c];
}

// softmax over length L, rows at stride ldr (padded score rows)
__global__ __launch_bounds__(256) void softmax_kernel(float* __restrict__ x,
                                                      int L, int ldr, float scale)
{
    __shared__ float red[256];
    float* row = x + (long)blockIdx.x * ldr;
    float m = -1e30f;
    for (int c = threadIdx.x; c < L; c += 256) m = fmaxf(m, row[c] * scale);
    red[threadIdx.x] = m; __syncthreads();
    for (int o = 128; o > 0; o >>= 1) {
        if (threadIdx.x < o) red[threadIdx.x] = fmaxf(red[threadIdx.x], red[threadIdx.x + o]);
        __syncthreads();
    }
    m = red[0]; __syncthreads();
    float s = 0.f;
    for (int c = threadIdx.x; c < L; c += 256) {
        float e = __expf(row[c] * scale - m); row[c] = e; s += e;
    }
    red[threadIdx.x] = s; __syncthreads();
    for (int o = 128; o > 0; o >>= 1) {
        if (threadIdx.x < o) red[threadIdx.x] += red[threadIdx.x + o];
        __syncthreads();
    }
    float inv = 1.f / red[0];
    for (int c = threadIdx.x; c < L; c += 256) row[c] *= inv;
}

__global__ void meanheads_kernel(const float* __restrict__ sc,
                                 float* __restrict__ out, int S, int ldr)
{
    long i = blockIdx.x * 256L + threadIdx.x;
    if (i >= (long)NB * S * S) return;
    int b = (int)(i / ((long)S * S));
    long r = i % ((long)S * S);
    int ri = (int)(r / S), ci = (int)(r % S);
    float s = 0.f;
    for (int h = 0; h < NHD; ++h)
        s += sc[((long)(b * NHD + h) * S + ri) * ldr + ci];
    out[i] = s * (1.f / NHD);
}

__global__ void pool_kernel(const float* __restrict__ a, float* __restrict__ out)
{
    int i = blockIdx.x * 256 + threadIdx.x;   // NB*16*16
    if (i >= NB * ADIM * ADIM) return;
    int b = i / (ADIM * ADIM), ij = i % (ADIM * ADIM);
    int pi = ij / ADIM, pj = ij % ADIM;
    int r0 = pi * SEQ / ADIM, r1 = ((pi + 1) * SEQ + ADIM - 1) / ADIM;
    int c0 = pj * SEQ / ADIM, c1 = ((pj + 1) * SEQ + ADIM - 1) / ADIM;
    float s = 0.f;
    for (int r = r0; r < r1; ++r)
        for (int c = c0; c < c1; ++c)
            s += a[((long)b * SEQ + r) * SEQ + c];
    out[i] = s / (float)((r1 - r0) * (c1 - c0));
}

__global__ void add_kernel(float* __restrict__ y, const float* __restrict__ x, long n)
{ long i = blockIdx.x * 256L + threadIdx.x; if (i < n) y[i] += x[i]; }

__global__ void copy_kernel(float* __restrict__ y, const float* __restrict__ x, long n)
{ long i = blockIdx.x * 256L + threadIdx.x; if (i < n) y[i] = x[i]; }

__global__ void fill_kernel(float* __restrict__ p, float v, long n)
{ long i = blockIdx.x * 256L + threadIdx.x; if (i < n) p[i] = v; }

__global__ void deg_kernel(const int* __restrict__ dst, float* __restrict__ deg, int E)
{ int e = blockIdx.x * 256 + threadIdx.x; if (e < E) atomicAdd(&deg[dst[e]], 1.f); }

__global__ void rsqrt_kernel(float* __restrict__ p, long n)
{ long i = blockIdx.x * 256L + threadIdx.x; if (i < n) p[i] = rsqrtf(p[i]); }

__global__ void gcn_self_kernel(const float* __restrict__ h, const float* __restrict__ dis,
                                const float* __restrict__ bias, float* __restrict__ out)
{
    long i = blockIdx.x * 256L + threadIdx.x;
    if (i >= (long)NNODES * DM) return;
    int node = (int)(i / DM), c = (int)(i % DM);
    float d = dis[node];
    out[i] = h[i] * d * d + bias[c];
}

__global__ __launch_bounds__(256) void gcn_scatter_kernel(
    const float* __restrict__ h, const float* __restrict__ dis,
    const int* __restrict__ src, const int* __restrict__ dst,
    float* __restrict__ out, int E)
{
    int e = blockIdx.x;
    if (e >= E) return;
    int s = src[e], d = dst[e];
    float w = dis[s] * dis[d];
    for (int c = threadIdx.x; c < DM; c += blockDim.x)
        atomicAdd(&out[(long)d * DM + c], h[(long)s * DM + c] * w);
}

__global__ void gather_rows_kernel(const float* __restrict__ feat, const int* __restrict__ idx,
                                   int idxStride, float* __restrict__ out, int nrows)
{
    long i = blockIdx.x * 256L + threadIdx.x;
    if (i >= (long)nrows * DM) return;
    int r = (int)(i / DM), c = (int)(i % DM);
    out[i] = feat[(long)idx[r * idxStride] * DM + c];
}

__global__ void scatter_rows_kernel(float* __restrict__ feat, const int* __restrict__ idx,
                                    int idxStride, const float* __restrict__ srcb,
                                    long srcStride, int nrows)
{
    long i = blockIdx.x * 256L + threadIdx.x;
    if (i >= (long)nrows * DM) return;
    int r = (int)(i / DM), c = (int)(i % DM);
    feat[(long)idx[r * idxStride] * DM + c] = srcb[(long)r * srcStride + c];
}

__global__ void concat2_kernel(const float* __restrict__ a, const float* __restrict__ b,
                               float* __restrict__ out, int per)
{
    int i = blockIdx.x * 256 + threadIdx.x;
    if (i >= NB * 2 * per) return;
    int row = i / (2 * per), c = i % (2 * per);
    out[i] = (c < per) ? a[row * per + c] : b[row * per + (c - per)];
}

__global__ void concat3_kernel(const float* __restrict__ a, const float* __restrict__ b,
                               const float* __restrict__ c, float* __restrict__ out)
{
    int i = blockIdx.x * 256 + threadIdx.x;
    if (i >= NB * DM) return;
    int row = i / DM, col = i % DM;
    float v = (col < 256) ? a[row * 256 + col]
            : (col < 512) ? b[row * 256 + (col - 256)]
                          : c[row * 256 + (col - 512)];
    out[i] = v;
}

// ---------------------------------------------------------------------------
// Host-side orchestration
// ---------------------------------------------------------------------------
static inline void launch_gemm(hipStream_t st,
    const float* A, long lda, long sA1, long sA2,
    const float* B, long ldb, long sB1, long sB2, int transB,
    float* C, long ldc, long sC1, long sC2,
    int M, int N, int K, const float* bias, int act, int nhdiv, int nz)
{
    dim3 grid((N + 63) / 64, (M + 63) / 64, nz);
    wmma_gemm_kernel<<<grid, dim3(128), 0, st>>>(A, lda, sA1, sA2,
        B, ldb, sB1, sB2, transB, C, ldc, sC1, sC2, M, N, K, bias, act, nhdiv);
}

static inline long cdiv(long a, long b) { return (a + b - 1) / b; }

// MHA: x (rows = NB*S2, DM) -> outb; scoresb ends softmaxed, row stride scp.
static void run_mha(hipStream_t st, const float* x, int S2, int scp,
                    const float* wqkv, const float* bqkv,
                    const float* wo, const float* bo,
                    float* qkvb, float* scoresb, float* attnob, float* outb)
{
    const float scale = 0.0721687836487032f;  // 1/sqrt(192)
    int M = NB * S2;
    launch_gemm(st, x, DM, 0, 0, wqkv, 3 * DM, 0, 0, 0,
                qkvb, 3 * DM, 0, 0, M, 3 * DM, DM, bqkv, 0, 1, 1);
    // scores[b,h] = q @ k^T   (row stride padded to scp for aligned reuse)
    launch_gemm(st, qkvb, 3 * DM, (long)S2 * 3 * DM, HD,
                qkvb + DM, 3 * DM, (long)S2 * 3 * DM, HD, 1,
                scoresb, scp, (long)NHD * S2 * scp, (long)S2 * scp,
                S2, S2, HD, nullptr, 0, NHD, NB * NHD);
    softmax_kernel<<<NB * NHD * S2, 256, 0, st>>>(scoresb, S2, scp, scale);
    // out[b,h] = attn @ v   (K = S2, zero-padded inside LDS)
    launch_gemm(st, scoresb, scp, (long)NHD * S2 * scp, (long)S2 * scp,
                qkvb + 2 * DM, 3 * DM, (long)S2 * 3 * DM, HD, 0,
                attnob, DM, (long)S2 * DM, HD,
                S2, HD, S2, nullptr, 0, NHD, NB * NHD);
    launch_gemm(st, attnob, DM, 0, 0, wo, DM, 0, 0, 0,
                outb, DM, 0, 0, M, DM, DM, bo, 0, 1, 1);
}

extern "C" void kernel_launch(void* const* d_in, const int* in_sizes, int n_in,
                              void* d_out, int out_size, void* d_ws, size_t ws_size,
                              hipStream_t stream)
{
    (void)in_sizes; (void)out_size; (void)ws_size;
    // Input order (setup_inputs dict order; params pytree flattened with
    // dict keys sorted alphabetically, tuples in order — JAX convention):
    //  0 text  1 nbr(int)  2 num  3 cat  4 tweet  5 des  6 edge_index(int)
    //  7 attn.W 8 attn.b 9 cat.W 10 cat.b 11 des.W 12 des.b 13 final.W 14 final.b
    // 15 i0.lg.W 16 i0.lg.b 17 i0.lt.W 18 i0.lt.b
    // 19..36 m0: gcn.W,gcn.b, lm.lin1.W/b, lm.lin2.W/b, lm.mha(wqkv,bqkv,wo,bo),
    //            lm.norm1(g,b), lm.norm2(g,b), mattn(wqkv,bqkv,wo,bo)
    // 37..54 m1 (same layout)
    // 55 num.W 56 num.b 57 title.W 58 title.b 59 tweet.W 60 tweet.b 61 user.W 62 user.b
    const float* P[64];
    for (int i = 0; i < n_in && i < 64; ++i) P[i] = (const float*)d_in[i];
    const float* text  = (const float*)d_in[0];
    const int*   nbr   = (const int*)d_in[1];
    const float* numf  = (const float*)d_in[2];
    const float* catf  = (const float*)d_in[3];
    const float* tweet = (const float*)d_in[4];
    const float* desf  = (const float*)d_in[5];
    const int*   edges = (const int*)d_in[6];
    const int* esrc = edges;
    const int* edst = edges + NEDGES;
    float* out = (float*)d_out;

    // ---- workspace layout (~330 MB of floats) ----
    float* ws = (float*)d_ws;
    size_t off = 0;
    auto alloc = [&](size_t n) { float* p = ws + off; off += (n + 255) & ~(size_t)255; return p; };
    float* feat  = alloc((size_t)NNODES * DM);
    float* hbuf  = alloc((size_t)NNODES * DM);   // GCN h; LM temps overlaid below
    float* xtext = alloc((size_t)NB * SEQ * DM);
    float* sub    = alloc((size_t)NB * KNBR * DM);
    float* subqkv = alloc((size_t)NB * KNBR * 3 * DM);
    float* subsc  = alloc((size_t)NB * NHD * KNBR * SUBSCP);
    float* subat  = alloc((size_t)NB * KNBR * DM);
    float* subout = alloc((size_t)NB * KNBR * DM);
    float* deg   = alloc(NNODES);
    float* g16   = alloc((size_t)NB * DM);
    float* t16   = alloc((size_t)NB * DM);
    float* ag0   = alloc(NB * ADIM * ADIM);
    float* ag1   = alloc(NB * ADIM * ADIM);
    float* title = alloc(NB * 256);
    float* uf    = alloc(NB * 256);
    float* av    = alloc(NB * 256);
    float* avin  = alloc(NB * 512);
    float* fin   = alloc((size_t)NB * DM);
    // LM temporaries overlaid on hbuf (LM runs before GCN in each _respective):
    float* qkv      = hbuf;                                     // NB*SEQ*3*DM
    float* scores   = qkv + (size_t)NB * SEQ * 3 * DM;          // NB*NHD*SEQ*SCP
    float* attno    = scores + (size_t)NB * NHD * SEQ * SCP;    // NB*SEQ*DM
    float* sabuf    = attno + (size_t)NB * SEQ * DM;            // NB*SEQ*DM
    float* lnbuf    = sabuf + (size_t)NB * SEQ * DM;            // NB*SEQ*DM
    float* meanattn = lnbuf + (size_t)NB * SEQ * DM;            // NB*SEQ*SEQ

    const long nText = (long)NB * SEQ * DM;

    // ---- 1. feature build: feat = concat(cat@, num@, tweet@, des@) ----
    smallk_gemm_kernel<<<cdiv((long)NNODES * 192, 256), 256, 0, stream>>>(
        catf, 3, P[9], P[10], feat + 0, DM, NNODES, 192, 3);
    smallk_gemm_kernel<<<cdiv((long)NNODES * 192, 256), 256, 0, stream>>>(
        numf, 5, P[55], P[56], feat + 192, DM, NNODES, 192, 5);
    launch_gemm(stream, tweet, DM, 0, 0, P[59], 192, 0, 0, 0,
                feat + 384, DM, 0, 0, NNODES, 192, DM, P[60], 0, 1, 1);
    launch_gemm(stream, desf, DM, 0, 0, P[11], 192, 0, 0, 0,
                feat + 576, DM, 0, 0, NNODES, 192, DM, P[12], 0, 1, 1);

    // ---- 2. copy text into working buffer ----
    copy_kernel<<<cdiv(nText, 256), 256, 0, stream>>>(xtext, text, nText);

    // ---- 3/5. two _respective blocks ----
    for (int layer = 0; layer < 2; ++layer) {
        const int base = (layer == 0) ? 19 : 37;
        float* ag = (layer == 0) ? ag0 : ag1;
        const float *gcnW = P[base+0], *gcnb = P[base+1];
        const float *l1W = P[base+2], *l1b = P[base+3];
        const float *l2W = P[base+4], *l2b = P[base+5];
        const float *wqkv = P[base+6], *bqkv = P[base+7];
        const float *wo = P[base+8], *bo = P[base+9];
        const float *n1g = P[base+10], *n1b = P[base+11];
        const float *n2g = P[base+12], *n2b = P[base+13];
        const float *mqkvW = P[base+14], *mqkvb = P[base+15];
        const float *mwo = P[base+16], *mbo = P[base+17];

        // --- transformer layer on text ---
        layernorm_kernel<<<NB * SEQ, 256, 0, stream>>>(xtext, n1g, n1b, lnbuf, DM);
        run_mha(stream, lnbuf, SEQ, SCP, wqkv, bqkv, wo, bo, qkv, scores, attno, sabuf);
        meanheads_kernel<<<cdiv((long)NB * SEQ * SEQ, 256), 256, 0, stream>>>(
            scores, meanattn, SEQ, SCP);
        pool_kernel<<<cdiv(NB * ADIM * ADIM, 256), 256, 0, stream>>>(meanattn, ag);
        add_kernel<<<cdiv(nText, 256), 256, 0, stream>>>(xtext, sabuf, nText);
        layernorm_kernel<<<NB * SEQ, 256, 0, stream>>>(xtext, n2g, n2b, lnbuf, DM);
        launch_gemm(stream, lnbuf, DM, 0, 0, l1W, DM, 0, 0, 0,
                    attno, DM, 0, 0, NB * SEQ, DM, DM, l1b, /*leaky*/1, 1, 1);
        launch_gemm(stream, attno, DM, 0, 0, l2W, DM, 0, 0, 0,
                    sabuf, DM, 0, 0, NB * SEQ, DM, DM, l2b, 0, 1, 1);
        add_kernel<<<cdiv(nText, 256), 256, 0, stream>>>(xtext, sabuf, nText);

        // --- GCN on feat (h -> hbuf, result back into feat) ---
        launch_gemm(stream, feat, DM, 0, 0, gcnW, DM, 0, 0, 0,
                    hbuf, DM, 0, 0, NNODES, DM, DM, nullptr, 0, 1, 1);
        fill_kernel<<<cdiv(NNODES, 256), 256, 0, stream>>>(deg, 1.f, NNODES);  // self loops
        deg_kernel<<<cdiv(NEDGES, 256), 256, 0, stream>>>(edst, deg, NEDGES);
        rsqrt_kernel<<<cdiv(NNODES, 256), 256, 0, stream>>>(deg, NNODES);      // deg -> dis
        gcn_self_kernel<<<cdiv((long)NNODES * DM, 256), 256, 0, stream>>>(hbuf, deg, gcnb, feat);
        gcn_scatter_kernel<<<NEDGES, 256, 0, stream>>>(hbuf, deg, esrc, edst, feat, NEDGES);

        // --- neighborhood MHA, write row-0 results back into feat ---
        gather_rows_kernel<<<cdiv((long)NB * KNBR * DM, 256), 256, 0, stream>>>(
            feat, nbr, 1, sub, NB * KNBR);
        run_mha(stream, sub, KNBR, SUBSCP, mqkvW, mqkvb, mwo, mbo,
                subqkv, subsc, subat, subout);
        scatter_rows_kernel<<<cdiv((long)NB * DM, 256), 256, 0, stream>>>(
            feat, nbr, KNBR, subout, (long)KNBR * DM, NB);

        // ---- 4. _interact (between the two layers only) ----
        if (layer == 0) {
            gather_rows_kernel<<<cdiv((long)NB * DM, 256), 256, 0, stream>>>(
                feat, nbr, KNBR, g16, NB);
            // text[:,0,:] = g @ lt  (write straight into token 0 via ldc = SEQ*DM)
            launch_gemm(stream, g16, DM, 0, 0, P[17], DM, 0, 0, 0,
                        xtext, (long)SEQ * DM, 0, 0, NB, DM, DM, P[18], 0, 1, 1);
            launch_gemm(stream, g16, DM, 0, 0, P[15], DM, 0, 0, 0,
                        t16, DM, 0, 0, NB, DM, DM, P[16], 0, 1, 1);
            scatter_rows_kernel<<<cdiv((long)NB * DM, 256), 256, 0, stream>>>(
                feat, nbr, KNBR, t16, DM, NB);
        }
    }

    // ---- 6. heads ----
    launch_gemm(stream, xtext, (long)SEQ * DM, 0, 0, P[57], 256, 0, 0, 0,
                title, 256, 0, 0, NB, 256, DM, P[58], 0, 1, 1);          // title(text[:,0])
    gather_rows_kernel<<<cdiv((long)NB * DM, 256), 256, 0, stream>>>(feat, nbr, KNBR, g16, NB);
    launch_gemm(stream, g16, DM, 0, 0, P[61], 256, 0, 0, 0,
                uf, 256, 0, 0, NB, 256, DM, P[62], 0, 1, 1);             // user(feat[ui])
    concat2_kernel<<<cdiv(NB * 512, 256), 256, 0, stream>>>(ag0, ag1, avin, 256);
    launch_gemm(stream, avin, 512, 0, 0, P[7], 256, 0, 0, 0,
                av, 256, 0, 0, NB, 256, 512, P[8], 0, 1, 1);             // attn head
    concat3_kernel<<<cdiv(NB * DM, 256), 256, 0, stream>>>(av, title, uf, fin);
    smallk_gemm_kernel<<<1, 64, 0, stream>>>(fin, DM, P[13], P[14], out, 2, NB, 2, DM);
}
